// DeepseekV2MLAAttention_29600914604512
// MI455X (gfx1250) — matrix operand order repaired and checked by reading, not compiled, not run on hardware
//
#include <hip/hip_runtime.h>
#include <math.h>

// ---------------------------------------------------------------------------
// DeepSeek-MLA forward for gfx1250 (MI455X): f16 WMMA GEMMs + indexer +
// radix-select top-k + flash attention with selection mask.
// ---------------------------------------------------------------------------

typedef __attribute__((ext_vector_type(16))) _Float16 v16h;
typedef __attribute__((ext_vector_type(8)))  _Float16 h8;
typedef __attribute__((ext_vector_type(8)))  float    v8f;

#define T_SEQ 2048
#define HID   2048
#define NHEAD 16
#define NOPE  128
#define ROPE  64
#define QKD   192
#define VD    128
#define QLR   1536
#define KVLR  512
#define IDXH  32
#define IDXD  128
#define TOPKN 512
#define QKVN  (QLR + KVLR + ROPE)   // 2112

static __device__ __forceinline__ v8f wmma16(v16h a, v16h b, v8f c) {
  // D = A(16x32 f16) x B(32x16 f16) + C(16x16 f32)
  return __builtin_amdgcn_wmma_f32_16x16x32_f16(false, a, false, b, (short)0, c,
                                                false, false);
}

// pack 8+8 consecutive f32 -> v16h (A/B fragment halves)
static __device__ __forceinline__ v16h pack16g(const float* p0, const float* p1) {
  union { v16h v; _Float16 h[16]; } u;
#pragma unroll
  for (int j = 0; j < 8; ++j) { u.h[j] = (_Float16)p0[j]; u.h[8 + j] = (_Float16)p1[j]; }
  return u.v;
}

// load 8+8 consecutive f16 from LDS (16B-aligned) -> v16h
static __device__ __forceinline__ v16h lds16(const _Float16* p0, const _Float16* p1) {
  union { v16h v; h8 h[2]; } u;
  u.h[0] = *(const h8*)p0;
  u.h[1] = *(const h8*)p1;
  return u.v;
}

static __device__ __forceinline__ float block_sum(float v, float* red, int nthreads) {
  const int lane = threadIdx.x & 31;
  const int wid  = threadIdx.x >> 5;
#pragma unroll
  for (int d = 16; d > 0; d >>= 1) v += __shfl_xor(v, d, 32);
  __syncthreads();
  if (lane == 0) red[wid] = v;
  __syncthreads();
  if (threadIdx.x == 0) {
    float s = 0.f;
    for (int i = 0; i < (nthreads >> 5); ++i) s += red[i];
    red[0] = s;
  }
  __syncthreads();
  return red[0];
}

static __device__ __forceinline__ void rope_rot(float pos, int p, int halfdim,
                                                float& x1, float& x2) {
  float inv = __powf(10000.0f, -(float)p / (float)halfdim);
  float f = pos * inv, sn, cs;
  __sincosf(f, &sn, &cs);
  float a = x1 * cs - x2 * sn;
  float b = x1 * sn + x2 * cs;
  x1 = a; x2 = b;
}

static __device__ __forceinline__ unsigned okey(float f) {
  unsigned u = __float_as_uint(f);
  return (u & 0x80000000u) ? ~u : (u | 0x80000000u);
}

// ---------------------------------------------------------------------------
// Generic GEMM: C[M,N] = A[M,K] @ B[K,N], f32 in/out, f16 WMMA inner product.
// 256 threads = 8 waves; block tile 128x128, wave tile 32x64, K-step 32.
// Requires M%128==0, K%32==0, N%4==0; N guarded (handles N=32..2112).
// ---------------------------------------------------------------------------
__global__ __launch_bounds__(256) void gemm_kernel(const float* __restrict__ A,
                                                   const float* __restrict__ B,
                                                   float* __restrict__ C,
                                                   int M, int N, int K) {
  __shared__ __align__(16) _Float16 Asl[128][40];   // [m][k]
  __shared__ __align__(16) _Float16 Bsl[128][40];   // [n][k] (transposed)
  const int tid  = threadIdx.x;
  const int wave = tid >> 5, lane = tid & 31;
  const int half = lane >> 4, l15 = lane & 15;
  const int bm = blockIdx.y * 128, bn = blockIdx.x * 128;
  const int wm = (wave >> 1) * 32, wn = (wave & 1) * 64;
  v8f acc[2][4] = {};
  for (int k0 = 0; k0 < K; k0 += 32) {
#pragma unroll
    for (int i = 0; i < 4; ++i) {               // stage A 128x32
      int r = (tid >> 3) + 32 * i;
      int c = (tid & 7) * 4;
      float4 v = *(const float4*)(A + (size_t)(bm + r) * K + k0 + c);
      Asl[r][c + 0] = (_Float16)v.x; Asl[r][c + 1] = (_Float16)v.y;
      Asl[r][c + 2] = (_Float16)v.z; Asl[r][c + 3] = (_Float16)v.w;
    }
#pragma unroll
    for (int i = 0; i < 4; ++i) {               // stage B 32x128 (transposed)
      int kk = (tid >> 5) + 8 * i;
      int n0 = (tid & 31) * 4;
      float4 v;
      if (bn + n0 < N) v = *(const float4*)(B + (size_t)(k0 + kk) * N + bn + n0);
      else { v.x = 0.f; v.y = 0.f; v.z = 0.f; v.w = 0.f; }
      Bsl[n0 + 0][kk] = (_Float16)v.x; Bsl[n0 + 1][kk] = (_Float16)v.y;
      Bsl[n0 + 2][kk] = (_Float16)v.z; Bsl[n0 + 3][kk] = (_Float16)v.w;
    }
    // prefetch next K panel into cache while this tile computes
    if (k0 + 32 < K) {
      const int r = (tid >> 3), c = (tid & 7) * 4;
      __builtin_prefetch(A + (size_t)(bm + r) * K + k0 + 32 + c, 0, 1);
      const int kk = (tid >> 5), n0 = (tid & 31) * 4;
      if (bn + n0 < N) __builtin_prefetch(B + (size_t)(k0 + 32 + kk) * N + bn + n0, 0, 1);
    }
    __syncthreads();
    v16h a[2], b[4];
#pragma unroll
    for (int r = 0; r < 2; ++r) {
      const _Float16* p = &Asl[wm + r * 16 + l15][0];
      a[r] = lds16(p + half * 8, p + 16 + half * 8);
    }
#pragma unroll
    for (int c = 0; c < 4; ++c) {
      const _Float16* p = &Bsl[wn + c * 16 + l15][0];
      b[c] = lds16(p + half * 8, p + 16 + half * 8);
    }
#pragma unroll
    for (int r = 0; r < 2; ++r)
#pragma unroll
      for (int c = 0; c < 4; ++c)
        acc[r][c] = wmma16(a[r], b[c], acc[r][c]);
    __syncthreads();
  }
#pragma unroll
  for (int r = 0; r < 2; ++r)
#pragma unroll
    for (int c = 0; c < 4; ++c)
#pragma unroll
      for (int i = 0; i < 8; ++i) {
        int row = bm + wm + r * 16 + half * 8 + i;
        int col = bn + wn + c * 16 + l15;
        if (row < M && col < N) C[(size_t)row * N + col] = acc[r][c][i];
      }
}

// ---------------------------------------------------------------------------
// RMS norms of q_c / kv_c + rope(k_pe)
// ---------------------------------------------------------------------------
__global__ __launch_bounds__(256) void norms_kernel(const float* __restrict__ qkv,
    const float* __restrict__ q_ln_w, const float* __restrict__ kv_ln_w,
    const int* __restrict__ positions,
    float* __restrict__ q_cn, float* __restrict__ kv_cn, float* __restrict__ k_pe_r) {
  const int t = blockIdx.x, tid = threadIdx.x;
  __shared__ float red[8];
  const float* row = qkv + (size_t)t * QKVN;
  float s = 0.f;
  for (int i = tid; i < QLR; i += 256) { float v = row[i]; s += v * v; }
  float inv = rsqrtf(block_sum(s, red, 256) * (1.0f / QLR) + 1e-6f);
  for (int i = tid; i < QLR; i += 256) q_cn[(size_t)t * QLR + i] = row[i] * inv * q_ln_w[i];
  s = 0.f;
  for (int i = tid; i < KVLR; i += 256) { float v = row[QLR + i]; s += v * v; }
  inv = rsqrtf(block_sum(s, red, 256) * (1.0f / KVLR) + 1e-6f);
  for (int i = tid; i < KVLR; i += 256)
    kv_cn[(size_t)t * KVLR + i] = row[QLR + i] * inv * kv_ln_w[i];
  float pos = (float)positions[t];
  if (tid < ROPE / 2) {
    float x1 = row[QLR + KVLR + 2 * tid], x2 = row[QLR + KVLR + 2 * tid + 1];
    rope_rot(pos, tid, ROPE / 2, x1, x2);
    k_pe_r[(size_t)t * ROPE + 2 * tid]     = x1;
    k_pe_r[(size_t)t * ROPE + 2 * tid + 1] = x2;
  }
}

// ---------------------------------------------------------------------------
// layer_norm(ki) + rope(ki_pe) in place, and scale w-projection
// ---------------------------------------------------------------------------
__global__ __launch_bounds__(128) void ln_w_kernel(float* __restrict__ ki,
    const float* __restrict__ gamma, const float* __restrict__ beta,
    const int* __restrict__ positions, float* __restrict__ wmat,
    const float* __restrict__ b_wproj) {
  const int t = blockIdx.x, tid = threadIdx.x;
  __shared__ float red[8];
  __shared__ float ybuf[IDXD];
  float x  = ki[(size_t)t * IDXD + tid];
  float mu = block_sum(x, red, 128) * (1.0f / IDXD);
  float d  = x - mu;
  float var = block_sum(d * d, red, 128) * (1.0f / IDXD);
  float y = d * rsqrtf(var + 1e-6f) * gamma[tid] + beta[tid];
  ybuf[tid] = y;
  __syncthreads();
  float pos = (float)positions[t];
  if (tid < ROPE / 2) {
    float x1 = ybuf[2 * tid], x2 = ybuf[2 * tid + 1];
    rope_rot(pos, tid, ROPE / 2, x1, x2);
    ki[(size_t)t * IDXD + 2 * tid]     = x1;
    ki[(size_t)t * IDXD + 2 * tid + 1] = x2;
  } else if (tid >= ROPE) {
    ki[(size_t)t * IDXD + tid] = ybuf[tid];
  }
  if (tid < IDXH)   // * IDX_D^-0.5 * IDX_H^-0.5 = 1/64
    wmat[(size_t)t * IDXH + tid] =
        (wmat[(size_t)t * IDXH + tid] + b_wproj[tid]) * (1.0f / 64.0f);
}

// rope first 64 dims of each of 32 index heads, in place
__global__ __launch_bounds__(256) void rope_qi_kernel(float* __restrict__ qi,
                                                      const int* __restrict__ positions) {
  const int t = blockIdx.x, tid = threadIdx.x;
  float pos = (float)positions[t];
  for (int pi = tid; pi < IDXH * (ROPE / 2); pi += 256) {
    int hh = pi >> 5, p = pi & 31;
    float* b = qi + (size_t)t * (IDXH * IDXD) + hh * IDXD + 2 * p;
    float x1 = b[0], x2 = b[1];
    rope_rot(pos, p, ROPE / 2, x1, x2);
    b[0] = x1; b[1] = x2;
  }
}

// rope q_pe in place + assemble k_full = [k_nope | k_pe_r]
__global__ __launch_bounds__(256) void assemble_kernel(float* __restrict__ q,
    const float* __restrict__ kv, const float* __restrict__ k_pe_r,
    const int* __restrict__ positions, float* __restrict__ k_full) {
  const int t = blockIdx.x, tid = threadIdx.x;
  float pos = (float)positions[t];
  for (int pi = tid; pi < NHEAD * (ROPE / 2); pi += 256) {
    int hh = pi >> 5, p = pi & 31;
    float* b = q + ((size_t)t * NHEAD + hh) * QKD + NOPE + 2 * p;
    float x1 = b[0], x2 = b[1];
    rope_rot(pos, p, ROPE / 2, x1, x2);
    b[0] = x1; b[1] = x2;
  }
  for (int i = tid; i < NHEAD * QKD; i += 256) {
    int hh = i / QKD, d = i % QKD;
    float v = (d < NOPE) ? kv[((size_t)t * NHEAD + hh) * (NOPE + VD) + d]
                         : k_pe_r[(size_t)t * ROPE + (d - NOPE)];
    k_full[((size_t)t * NHEAD + hh) * QKD + d] = v;
  }
}

// ---------------------------------------------------------------------------
// Indexer score: score[t,s] = sum_h w[t,h] * relu(qi[t,h,:] . ki[s,:])
// One wave per 16x16 (t,s) tile; lower triangle tiles only.
// ---------------------------------------------------------------------------
__global__ __launch_bounds__(128) void score_kernel(const float* __restrict__ qi,
    const float* __restrict__ ki, const float* __restrict__ wmat,
    float* __restrict__ score) {
  const int tb = blockIdx.x * 16;
  const int sg = blockIdx.y * 64;
  if (sg > tb + 15) return;
  const int wave = threadIdx.x >> 5, lane = threadIdx.x & 31;
  const int sb = sg + wave * 16;
  if (sb > tb + 15) return;
  const int half = lane >> 4, l15 = lane & 15;
  v16h bf[4];
  {
    const float* kp = ki + (size_t)(sb + l15) * IDXD;
#pragma unroll
    for (int c = 0; c < 4; ++c)
      bf[c] = pack16g(kp + c * 32 + half * 8, kp + c * 32 + 16 + half * 8);
  }
  v8f acc = {};
  const float* qrow = qi + (size_t)(tb + l15) * (IDXH * IDXD);
  for (int h = 0; h < IDXH; ++h) {
    v8f lacc = {};
#pragma unroll
    for (int c = 0; c < 4; ++c) {
      const float* p = qrow + h * IDXD + c * 32;
      v16h a = pack16g(p + half * 8, p + 16 + half * 8);
      lacc = wmma16(a, bf[c], lacc);
    }
#pragma unroll
    for (int i = 0; i < 8; ++i) {
      float v = lacc[i];
      v = v > 0.f ? v : 0.f;
      acc[i] += v * wmat[(size_t)(tb + half * 8 + i) * IDXH + h];
    }
  }
#pragma unroll
  for (int i = 0; i < 8; ++i)
    score[(size_t)(tb + half * 8 + i) * T_SEQ + sb + l15] = acc[i];
}

// ---------------------------------------------------------------------------
// Per-row top-512 via 4-pass MSD radix select -> selection bitmask
// ---------------------------------------------------------------------------
__global__ __launch_bounds__(256) void topk_kernel(const float* __restrict__ score,
                                                   unsigned* __restrict__ maskw) {
  const int t = blockIdx.x, tid = threadIdx.x;
  const int n = t + 1;
  __shared__ int hist[256];
  __shared__ unsigned sh_pref;
  __shared__ int sh_k;
  unsigned thr = 0;
  const bool all = (n <= TOPKN);
  if (!all) {
    unsigned prefix = 0;
    int k = TOPKN;
    for (int pass = 0; pass < 4; ++pass) {
      hist[tid] = 0;
      __syncthreads();
      const int shift = 24 - 8 * pass;
      for (int s = tid; s < n; s += 256) {
        unsigned key = okey(score[(size_t)t * T_SEQ + s]);
        bool cand = (pass == 0) || ((key >> (shift + 8)) == prefix);
        if (cand) atomicAdd(&hist[(key >> shift) & 255], 1);
      }
      __syncthreads();
      if (tid == 0) {
        int cum = 0, b = 255;
        for (; b > 0; --b) { cum += hist[b]; if (cum >= k) break; }
        if (cum < k) { cum += hist[0]; b = 0; }
        sh_k = k - (cum - hist[b]);
        sh_pref = (prefix << 8) | (unsigned)b;
      }
      __syncthreads();
      prefix = sh_pref;
      k = sh_k;
      __syncthreads();
    }
    thr = prefix;
  }
  for (int w = tid; w < T_SEQ / 32; w += 256) {
    unsigned bits = 0;
    for (int j = 0; j < 32; ++j) {
      int s = w * 32 + j;
      if (s <= t && (all || okey(score[(size_t)t * T_SEQ + s]) >= thr))
        bits |= (1u << j);
    }
    maskw[(size_t)t * (T_SEQ / 32) + w] = bits;
  }
}

// ---------------------------------------------------------------------------
// Masked flash attention. Block = (64 q-rows, 1 head); 4 waves x 16 rows.
// s-tiles of 32; K/V staged to LDS in f16; online softmax.
// ---------------------------------------------------------------------------
__global__ __launch_bounds__(128) void attn_kernel(const float* __restrict__ q,
    const float* __restrict__ k_full, const float* __restrict__ kv,
    const unsigned* __restrict__ maskw, float* __restrict__ attn_out) {
  const float SCALING = 0.07216878364870323f;   // 192^-0.5
  const int h    = blockIdx.y;
  const int btb  = blockIdx.x * 64;
  const int tid  = threadIdx.x;
  const int wave = tid >> 5, lane = tid & 31;
  const int half = lane >> 4, l15 = lane & 15;
  const int tbw  = btb + wave * 16;

  __shared__ __align__(16) _Float16 Kls[32][208];     // [s][d]
  __shared__ __align__(16) _Float16 Vls[128][40];     // [d][s] (transposed)
  __shared__ __align__(16) _Float16 Pls[4][16][40];   // per-wave P tile
  __shared__ unsigned Mls[64];

  v16h qf[6];
  {
    const float* qrow = q + ((size_t)(tbw + l15) * NHEAD + h) * QKD;
#pragma unroll
    for (int c = 0; c < 6; ++c)
      qf[c] = pack16g(qrow + c * 32 + half * 8, qrow + c * 32 + 16 + half * 8);
  }
  v8f o[8] = {};
  float m_i[8], l_i[8];
#pragma unroll
  for (int i = 0; i < 8; ++i) { m_i[i] = -1e30f; l_i[i] = 0.f; }

  const int tmax = btb + 63;
  for (int s0 = 0; s0 <= tmax; s0 += 32) {
    {   // stage K tile (32 x 192) f32 -> f16
      const int r  = tid >> 2;
      const int d0 = (tid & 3) * 48;
      const float* src = k_full + ((size_t)(s0 + r) * NHEAD + h) * QKD + d0;
#pragma unroll
      for (int j = 0; j < 48; j += 4) {
        float4 v = *(const float4*)(src + j);
        Kls[r][d0 + j + 0] = (_Float16)v.x; Kls[r][d0 + j + 1] = (_Float16)v.y;
        Kls[r][d0 + j + 2] = (_Float16)v.z; Kls[r][d0 + j + 3] = (_Float16)v.w;
      }
    }
    {   // stage V tile (32 x 128) transposed
      const int r  = tid >> 2;
      const int d0 = (tid & 3) * 32;
      const float* src = kv + ((size_t)(s0 + r) * NHEAD + h) * (NOPE + VD) + NOPE + d0;
#pragma unroll
      for (int j = 0; j < 32; j += 4) {
        float4 v = *(const float4*)(src + j);
        Vls[d0 + j + 0][r] = (_Float16)v.x; Vls[d0 + j + 1][r] = (_Float16)v.y;
        Vls[d0 + j + 2][r] = (_Float16)v.z; Vls[d0 + j + 3][r] = (_Float16)v.w;
      }
    }
    if (tid < 64) Mls[tid] = maskw[(size_t)(btb + tid) * (T_SEQ / 32) + (s0 >> 5)];
    __syncthreads();

    // Q @ K^T for this s-tile (two 16-wide column groups)
    v8f sc[2] = {};
#pragma unroll
    for (int c = 0; c < 6; ++c) {
#pragma unroll
      for (int sub = 0; sub < 2; ++sub) {
        const _Float16* p = &Kls[sub * 16 + l15][c * 32 + half * 8];
        v16h b = lds16(p, p + 16);
        sc[sub] = wmma16(qf[c], b, sc[sub]);
      }
    }

    // online softmax update per owned row
#pragma unroll
    for (int i = 0; i < 8; ++i) {
      const int rrow = half * 8 + i;
      const int t    = tbw + rrow;
      const unsigned mw = Mls[(tbw - btb) + rrow];
      const bool sel0 = (s0 + l15 <= t)      && ((mw >> l15) & 1);
      const bool sel1 = (s0 + 16 + l15 <= t) && ((mw >> (16 + l15)) & 1);
      float v0 = sel0 ? sc[0][i] * SCALING : -1e30f;
      float v1 = sel1 ? sc[1][i] * SCALING : -1e30f;
      float tm = fmaxf(v0, v1);
#pragma unroll
      for (int d2 = 1; d2 < 16; d2 <<= 1) tm = fmaxf(tm, __shfl_xor(tm, d2, 32));
      float mnew  = fmaxf(m_i[i], tm);
      float alpha = __expf(m_i[i] - mnew);
      float p0 = sel0 ? __expf(v0 - mnew) : 0.f;
      float p1 = sel1 ? __expf(v1 - mnew) : 0.f;
      float ps = p0 + p1;
#pragma unroll
      for (int d2 = 1; d2 < 16; d2 <<= 1) ps += __shfl_xor(ps, d2, 32);
      l_i[i] = l_i[i] * alpha + ps;
      m_i[i] = mnew;
#pragma unroll
      for (int nb = 0; nb < 8; ++nb) o[nb][i] *= alpha;
      Pls[wave][rrow][l15]      = (_Float16)p0;
      Pls[wave][rrow][16 + l15] = (_Float16)p1;
    }

    // P @ V
    {
      const _Float16* pp = &Pls[wave][l15][0];
      v16h pa = lds16(pp + half * 8, pp + 16 + half * 8);
#pragma unroll
      for (int nb = 0; nb < 8; ++nb) {
        const _Float16* vp = &Vls[nb * 16 + l15][half * 8];
        v16h vb = lds16(vp, vp + 16);
        o[nb] = wmma16(pa, vb, o[nb]);
      }
    }
    __syncthreads();
  }

#pragma unroll
  for (int i = 0; i < 8; ++i) {
    const float inv = 1.0f / l_i[i];
    const int t = tbw + half * 8 + i;
#pragma unroll
    for (int nb = 0; nb < 8; ++nb)
      attn_out[(size_t)t * HID + h * VD + nb * 16 + l15] = o[nb][i] * inv;
  }
}

// ---------------------------------------------------------------------------
extern "C" void kernel_launch(void* const* d_in, const int* in_sizes, int n_in,
                              void* d_out, int out_size, void* d_ws, size_t ws_size,
                              hipStream_t stream) {
  const int*   positions = (const int*)  d_in[0];
  const float* hidden    = (const float*)d_in[1];
  const float* w_qkv_a   = (const float*)d_in[2];
  const float* q_a_ln_w  = (const float*)d_in[3];
  const float* w_qb      = (const float*)d_in[4];
  const float* kv_a_ln_w = (const float*)d_in[5];
  const float* w_kvb     = (const float*)d_in[6];
  const float* w_o       = (const float*)d_in[7];
  const float* w_idx_qb  = (const float*)d_in[8];
  const float* w_idx_k   = (const float*)d_in[9];
  const float* gamma     = (const float*)d_in[10];
  const float* beta      = (const float*)d_in[11];
  const float* w_wproj   = (const float*)d_in[12];
  const float* b_wproj   = (const float*)d_in[13];
  float* out = (float*)d_out;

  float* p = (float*)d_ws;
  float* qkv    = p; p += (size_t)T_SEQ * QKVN;
  float* q_cn   = p; p += (size_t)T_SEQ * QLR;
  float* kv_cn  = p; p += (size_t)T_SEQ * KVLR;
  float* k_pe_r = p; p += (size_t)T_SEQ * ROPE;
  float* qi     = p; p += (size_t)T_SEQ * IDXH * IDXD;
  float* ki     = p; p += (size_t)T_SEQ * IDXD;
  float* wmat   = p; p += (size_t)T_SEQ * IDXH;
  float* score  = p; p += (size_t)T_SEQ * T_SEQ;
  unsigned* maskb = (unsigned*)p; p += (size_t)T_SEQ * (T_SEQ / 32);
  float* qbuf   = p; p += (size_t)T_SEQ * NHEAD * QKD;
  float* kvbuf  = p; p += (size_t)T_SEQ * NHEAD * (NOPE + VD);
  float* kfull  = p; p += (size_t)T_SEQ * NHEAD * QKD;
  float* attn_o = p; p += (size_t)T_SEQ * HID;

  const dim3 gblk(256);
  const dim3 blk(128);
#define GX(n) (((n) + 127) / 128)
  gemm_kernel<<<dim3(GX(QKVN), T_SEQ / 128), gblk, 0, stream>>>(hidden, w_qkv_a, qkv, T_SEQ, QKVN, HID);
  norms_kernel<<<T_SEQ, 256, 0, stream>>>(qkv, q_a_ln_w, kv_a_ln_w, positions, q_cn, kv_cn, k_pe_r);
  gemm_kernel<<<dim3(GX(IDXH * IDXD), T_SEQ / 128), gblk, 0, stream>>>(q_cn, w_idx_qb, qi, T_SEQ, IDXH * IDXD, QLR);
  gemm_kernel<<<dim3(GX(IDXD), T_SEQ / 128), gblk, 0, stream>>>(hidden, w_idx_k, ki, T_SEQ, IDXD, HID);
  gemm_kernel<<<dim3(GX(IDXH), T_SEQ / 128), gblk, 0, stream>>>(hidden, w_wproj, wmat, T_SEQ, IDXH, HID);
  ln_w_kernel<<<T_SEQ, 128, 0, stream>>>(ki, gamma, beta, positions, wmat, b_wproj);
  rope_qi_kernel<<<T_SEQ, 256, 0, stream>>>(qi, positions);
  score_kernel<<<dim3(T_SEQ / 16, T_SEQ / 64), blk, 0, stream>>>(qi, ki, wmat, score);
  topk_kernel<<<T_SEQ, 256, 0, stream>>>(score, maskb);
  gemm_kernel<<<dim3(GX(NHEAD * QKD), T_SEQ / 128), gblk, 0, stream>>>(q_cn, w_qb, qbuf, T_SEQ, NHEAD * QKD, QLR);
  gemm_kernel<<<dim3(GX(NHEAD * (NOPE + VD)), T_SEQ / 128), gblk, 0, stream>>>(kv_cn, w_kvb, kvbuf, T_SEQ, NHEAD * (NOPE + VD), KVLR);
  assemble_kernel<<<T_SEQ, 256, 0, stream>>>(qbuf, kvbuf, k_pe_r, positions, kfull);
  attn_kernel<<<dim3(T_SEQ / 64, NHEAD), blk, 0, stream>>>(qbuf, kfull, kvbuf, maskb, attn_o);
  gemm_kernel<<<dim3(GX(HID), T_SEQ / 128), gblk, 0, stream>>>(attn_o, w_o, out, T_SEQ, HID, HID);
#undef GX

  (void)in_sizes; (void)n_in; (void)out_size; (void)ws_size;
}
